// GAT2LayersTFM_56624848830742
// MI455X (gfx1250) — matrix operand order repaired and checked by reading, not compiled, not run on hardware
//
#include <hip/hip_runtime.h>

#define NN 100000
#define DD 128
#define HH 4
#define CC 32
#define FF 512
#define EE 600000

typedef __attribute__((ext_vector_type(8)))  _Float16 v8h;
typedef __attribute__((ext_vector_type(16))) _Float16 v16h;
typedef __attribute__((ext_vector_type(8)))  float    v8f;

// ---- order-preserving float <-> uint encoding for atomicMax-based segment max ----
__device__ __forceinline__ unsigned encOrd(float f) {
  unsigned u = __float_as_uint(f);
  return (u & 0x80000000u) ? ~u : (u | 0x80000000u);
}
__device__ __forceinline__ float decOrd(unsigned u) {
  return (u & 0x80000000u) ? __uint_as_float(u ^ 0x80000000u) : __uint_as_float(~u);
}

// ---- pack fp32 weight W[K x Nc] (row-major) into WMMA B-fragment order (f16) ----
// PB[((kt*ntn + nt)*32 + L)*16 + j] = W[kt*32 + (j&7) + 8*(L>>4) + 16*(j>>3)][nt*16 + (L&15)]
__global__ void gat_pack_b_kernel(const float* __restrict__ W, _Float16* __restrict__ PB,
                                  int K, int Nc) {
  int t = blockIdx.x * blockDim.x + threadIdx.x;
  if (t >= K * Nc) return;
  int j    = t & 15;
  int L    = (t >> 4) & 31;
  int rest = t >> 9;
  int ntn  = Nc >> 4;
  int nt   = rest % ntn;
  int kt   = rest / ntn;
  int k = kt * 32 + (j & 7) + ((L >> 4) << 3) + ((j >> 3) << 4);
  int n = nt * 16 + (L & 15);
  PB[t] = (_Float16)W[(size_t)k * Nc + n];
}

__global__ void gat_f32_to_f16_kernel(const float* __restrict__ in, _Float16* __restrict__ out,
                                      int n) {
  int t = blockIdx.x * blockDim.x + threadIdx.x;
  if (t < n) out[t] = (_Float16)in[t];
}

// ---- WMMA GEMM: out[MxNc] = A[MxK] @ W + bias (+resid) (+relu); A f16 row-major, W pre-packed ----
__global__ void gat_wmma_gemm_kernel(const _Float16* __restrict__ A, const _Float16* __restrict__ PB,
                                     const float* __restrict__ bias, const float* __restrict__ resid,
                                     float* __restrict__ outF, _Float16* __restrict__ outH,
                                     int K, int Nc, int relu) {
  int lane = threadIdx.x;                 // 0..31 (wave32)
  int tn = blockIdx.x, tm = blockIdx.y;   // 16x16 output tile
  int row = tm * 16 + (lane & 15);        // A row for this lane
  int hi  = lane >> 4;                    // lane-half selects K sub-block
  v8f acc = {};
  int ktiles = K >> 5;
  int ntn = Nc >> 4;
  const _Float16* ap = A + (size_t)row * K + hi * 8;
  const _Float16* bp = PB + ((size_t)tn * 32 + lane) * 16;
  size_t bstep = (size_t)ntn * 512;       // halfs per K-tile of packed B
  for (int kt = 0; kt < ktiles; ++kt) {
    v8h a0 = *(const v8h*)(ap);           // K = kt*32 + 8*hi + [0..7]
    v8h a1 = *(const v8h*)(ap + 16);      // K = kt*32 + 16 + 8*hi + [0..7]
    v16h av = __builtin_shufflevector(a0, a1, 0,1,2,3,4,5,6,7,8,9,10,11,12,13,14,15);
    v16h bv = *(const v16h*)(bp);
    acc = __builtin_amdgcn_wmma_f32_16x16x32_f16(false, av, false, bv,
                                                 (short)0, acc, false, false);
    ap += 32;
    bp += bstep;
  }
  int col = tn * 16 + (lane & 15);
  float bb = bias ? bias[col] : 0.0f;
#pragma unroll
  for (int r = 0; r < 8; ++r) {
    int m = tm * 16 + r + 8 * hi;
    size_t idx = (size_t)m * Nc + col;
    float v = acc[r] + bb;
    if (resid) v += resid[idx];
    if (relu)  v = fmaxf(v, 0.0f);
    if (outF) outF[idx] = v;
    if (outH) outH[idx] = (_Float16)v;
  }
}

// ---- GAT per-layer init: out = bias (broadcast), amax/denom cleared ----
__global__ void gat_init_kernel(const float* __restrict__ bias, float* __restrict__ g,
                                unsigned* __restrict__ amax, float* __restrict__ denom) {
  int t = blockIdx.x * blockDim.x + threadIdx.x;
  if (t >= NN * DD) return;
  int i = t & (DD - 1);
  g[t] = bias[i];
  if (i < HH) {
    int n = t >> 7;
    amax[n * HH + i]  = 0u;     // encOrd-minimum (acts as -inf)
    denom[n * HH + i] = 0.0f;
  }
}

// ---- edge kernel 1: alpha[e,h] = sum_c att[h,c]*leaky(xl[src]+xr[dst]); segment max ----
__global__ void gat_edge_alpha_kernel(const int* __restrict__ src, const int* __restrict__ dst,
                                      const float* __restrict__ xl, const float* __restrict__ xr,
                                      const float* __restrict__ att,
                                      float* __restrict__ alpha, unsigned* __restrict__ amax) {
  int t = blockIdx.x * blockDim.x + threadIdx.x;
  if (t >= EE * HH) return;
  int e = t >> 2, h = t & 3;
  int s = src[e], d = dst[e];
  const float4* pl = (const float4*)(xl + (size_t)s * DD + h * CC);
  const float4* pr = (const float4*)(xr + (size_t)d * DD + h * CC);
  const float4* pa = (const float4*)(att + h * CC);
  float acc = 0.0f;
#pragma unroll
  for (int i = 0; i < CC / 4; ++i) {
    float4 a = pl[i], b = pr[i], w = pa[i];
    float e0 = a.x + b.x; e0 = e0 > 0.0f ? e0 : 0.2f * e0;
    float e1 = a.y + b.y; e1 = e1 > 0.0f ? e1 : 0.2f * e1;
    float e2 = a.z + b.z; e2 = e2 > 0.0f ? e2 : 0.2f * e2;
    float e3 = a.w + b.w; e3 = e3 > 0.0f ? e3 : 0.2f * e3;
    acc += e0 * w.x + e1 * w.y + e2 * w.z + e3 * w.w;
  }
  alpha[t] = acc;
  atomicMax(&amax[(size_t)d * HH + h], encOrd(acc));
}

// ---- edge kernel 2: ex = exp(alpha - amax[dst]); denom[dst] += ex ----
__global__ void gat_edge_exp_kernel(const int* __restrict__ dst, float* __restrict__ alpha,
                                    const unsigned* __restrict__ amax, float* __restrict__ denom) {
  int t = blockIdx.x * blockDim.x + threadIdx.x;
  if (t >= EE * HH) return;
  int e = t >> 2, h = t & 3;
  int d = dst[e];
  float m = decOrd(amax[(size_t)d * HH + h]);
  float ex = __expf(alpha[t] - m);
  alpha[t] = ex;
  atomicAdd(&denom[(size_t)d * HH + h], ex);
}

// ---- edge kernel 3: out[dst,h,:] += xl[src,h,:] * (ex/denom[dst,h]) ----
__global__ void gat_edge_aggr_kernel(const int* __restrict__ src, const int* __restrict__ dst,
                                     const float* __restrict__ xl, const float* __restrict__ alpha,
                                     const float* __restrict__ denom, float* __restrict__ g) {
  int t = blockIdx.x * blockDim.x + threadIdx.x;
  if (t >= EE * HH) return;
  int e = t >> 2, h = t & 3;
  int s = src[e], d = dst[e];
  float a = alpha[t] / (denom[(size_t)d * HH + h] + 1e-16f);
  const float4* pl = (const float4*)(xl + (size_t)s * DD + h * CC);
  float* po = g + (size_t)d * DD + h * CC;
#pragma unroll
  for (int i = 0; i < CC / 4; ++i) {
    float4 v = pl[i];
    atomicAdd(po + 4 * i + 0, v.x * a);
    atomicAdd(po + 4 * i + 1, v.y * a);
    atomicAdd(po + 4 * i + 2, v.z * a);
    atomicAdd(po + 4 * i + 3, v.w * a);
  }
}

// ---- LayerNorm: one wave32 per row of 128; emits fp32 (and optional f16 copy) ----
__global__ void gat_ln_kernel(const float* __restrict__ y, const float* __restrict__ gg,
                              const float* __restrict__ bb, float* __restrict__ outF,
                              _Float16* __restrict__ outH) {
  int lane = threadIdx.x & 31;
  int wid  = threadIdx.x >> 5;
  int row  = blockIdx.x * 8 + wid;
  if (row >= NN) return;
  const float4* p = (const float4*)(y + (size_t)row * DD);
  float4 v = p[lane];
  float s = v.x + v.y + v.z + v.w;
  float q = v.x * v.x + v.y * v.y + v.z * v.z + v.w * v.w;
#pragma unroll
  for (int m = 16; m >= 1; m >>= 1) {
    s += __shfl_xor(s, m, 32);
    q += __shfl_xor(q, m, 32);
  }
  float mean = s * (1.0f / DD);
  float var  = q * (1.0f / DD) - mean * mean;
  float rstd = rsqrtf(var + 1e-5f);
  int i0 = lane * 4;
  float o0 = (v.x - mean) * rstd * gg[i0 + 0] + bb[i0 + 0];
  float o1 = (v.y - mean) * rstd * gg[i0 + 1] + bb[i0 + 1];
  float o2 = (v.z - mean) * rstd * gg[i0 + 2] + bb[i0 + 2];
  float o3 = (v.w - mean) * rstd * gg[i0 + 3] + bb[i0 + 3];
  float4 o; o.x = o0; o.y = o1; o.z = o2; o.w = o3;
  ((float4*)(outF + (size_t)row * DD))[lane] = o;
  if (outH) {
    _Float16* ph = outH + (size_t)row * DD + i0;
    ph[0] = (_Float16)o0; ph[1] = (_Float16)o1;
    ph[2] = (_Float16)o2; ph[3] = (_Float16)o3;
  }
}

extern "C" void kernel_launch(void* const* d_in, const int* in_sizes, int n_in,
                              void* d_out, int out_size, void* d_ws, size_t ws_size,
                              hipStream_t stream) {
  (void)in_sizes; (void)n_in; (void)out_size; (void)ws_size;
  const float* x  = (const float*)d_in[0];
  const int*   ei = (const int*)d_in[1];
  const float* cw[2][6];  // wl, bl, wr, br, att, bias
  for (int p = 0; p < 2; ++p)
    for (int k = 0; k < 6; ++k) cw[p][k] = (const float*)d_in[2 + p * 6 + k];
  const float* fw[2][4];  // w1, b1, w2, b2
  for (int p = 0; p < 2; ++p)
    for (int k = 0; k < 4; ++k) fw[p][k] = (const float*)d_in[14 + p * 4 + k];
  const float* lng[2] = {(const float*)d_in[22], (const float*)d_in[24]};
  const float* lnb[2] = {(const float*)d_in[23], (const float*)d_in[25]};

  char* ws = (char*)d_ws;
  size_t off = 0;
  auto alloc = [&](size_t bytes) -> char* {
    char* p = ws + off;
    off += (bytes + 255) & ~(size_t)255;
    return p;
  };
  float*     xl    = (float*)alloc((size_t)NN * DD * 4);  // xl; later reused as FFN output y
  float*     xr    = (float*)alloc((size_t)NN * DD * 4);  // xr; later LN fp32 scratch
  float*     g     = (float*)alloc((size_t)NN * DD * 4);  // GAT output / FFN input / residual
  _Float16*  tbuf  = (_Float16*)alloc((size_t)NN * FF * 2);
  _Float16*  h16   = (_Float16*)alloc((size_t)NN * DD * 2);  // f16 layer input
  _Float16*  g16   = (_Float16*)alloc((size_t)NN * DD * 2);  // f16 FFN input
  float*     alpha = (float*)alloc((size_t)EE * HH * 4);
  unsigned*  amax  = (unsigned*)alloc((size_t)NN * HH * 4);
  float*     denom = (float*)alloc((size_t)NN * HH * 4);
  _Float16*  pb_cl[2]; _Float16* pb_cr[2]; _Float16* pb_w1[2]; _Float16* pb_w2[2];
  for (int p = 0; p < 2; ++p) {
    pb_cl[p] = (_Float16*)alloc((size_t)DD * DD * 2);
    pb_cr[p] = (_Float16*)alloc((size_t)DD * DD * 2);
    pb_w1[p] = (_Float16*)alloc((size_t)DD * FF * 2);
    pb_w2[p] = (_Float16*)alloc((size_t)FF * DD * 2);
  }

  // pre-pack all weights into WMMA B-fragment order (f16)
  for (int p = 0; p < 2; ++p) {
    gat_pack_b_kernel<<<(DD * DD + 255) / 256, 256, 0, stream>>>(cw[p][0], pb_cl[p], DD, DD);
    gat_pack_b_kernel<<<(DD * DD + 255) / 256, 256, 0, stream>>>(cw[p][2], pb_cr[p], DD, DD);
    gat_pack_b_kernel<<<(DD * FF + 255) / 256, 256, 0, stream>>>(fw[p][0], pb_w1[p], DD, FF);
    gat_pack_b_kernel<<<(FF * DD + 255) / 256, 256, 0, stream>>>(fw[p][2], pb_w2[p], FF, DD);
  }
  gat_f32_to_f16_kernel<<<(NN * DD + 255) / 256, 256, 0, stream>>>(x, h16, NN * DD);

  const int* srcp = ei;
  const int* dstp = ei + EE;
  int eblocks = (EE * HH + 255) / 256;

  for (int p = 0; p < 2; ++p) {
    // GATv2 linear transforms (WMMA)
    dim3 gl(DD / 16, NN / 16);
    gat_wmma_gemm_kernel<<<gl, 32, 0, stream>>>(h16, pb_cl[p], cw[p][1], nullptr,
                                                xl, nullptr, DD, DD, 0);
    gat_wmma_gemm_kernel<<<gl, 32, 0, stream>>>(h16, pb_cr[p], cw[p][3], nullptr,
                                                xr, nullptr, DD, DD, 0);
    // segment softmax + aggregation over edges
    gat_init_kernel<<<(NN * DD + 255) / 256, 256, 0, stream>>>(cw[p][5], g, amax, denom);
    gat_edge_alpha_kernel<<<eblocks, 256, 0, stream>>>(srcp, dstp, xl, xr, cw[p][4], alpha, amax);
    gat_edge_exp_kernel<<<eblocks, 256, 0, stream>>>(dstp, alpha, amax, denom);
    gat_edge_aggr_kernel<<<eblocks, 256, 0, stream>>>(srcp, dstp, xl, alpha, denom, g);
    // FFN (WMMA) + residual
    gat_f32_to_f16_kernel<<<(NN * DD + 255) / 256, 256, 0, stream>>>(g, g16, NN * DD);
    dim3 gf1(FF / 16, NN / 16);
    gat_wmma_gemm_kernel<<<gf1, 32, 0, stream>>>(g16, pb_w1[p], fw[p][1], nullptr,
                                                 nullptr, tbuf, DD, FF, 1);
    dim3 gf2(DD / 16, NN / 16);
    gat_wmma_gemm_kernel<<<gf2, 32, 0, stream>>>(tbuf, pb_w2[p], fw[p][3], g,
                                                 xl, nullptr, FF, DD, 0);
    // LayerNorm; layer 0 feeds layer 1 (f16), layer 1 writes final fp32 output
    float*    lnF = (p == 0) ? xr : (float*)d_out;
    _Float16* lnH = (p == 0) ? h16 : nullptr;
    gat_ln_kernel<<<NN / 8, 256, 0, stream>>>(xl, lng[p], lnb[p], lnF, lnH);
  }
}
